// PhysicsAttentionV3_73143293051546
// MI455X (gfx1250) — matrix operand order repaired and checked by reading, not compiled, not run on hardware
//
#include <hip/hip_runtime.h>
#include <hip/hip_bf16.h>

// PhysicsAttentionV3 for MI455X (gfx1250, wave32, WMMA bf16).
// Pipeline:
//   k_init : zero d (H*M) + s_raw (H*M*C); convert w_slice -> bf16 (L2-resident)
//   k1     : logits = x @ w_slice^T (WMMA bf16), +bias, /temp, softmax over M
//            per head; atomicAdd d; store w as bf16 [n][h*64+m] (102.4 MB)
//   k2     : s_raw[h][m][c] += sum_n w[h,n,m]*x[n,c]  (split-K over node
//            chunks; head is the FASTEST grid dim so all 8 heads + 8 c-blocks
//            of a chunk share its x/w tiles in L2; LDS-transposed bf16 tiles
//            with padded stride 130 (65 dwords, conflict-free); WMMA; f32
//            atomics resolved in L2; global_prefetch_b8 for next stage)
//   k3a    : s = (s_raw/d) @ w1^T + b1     (tiny, plain VALU)
//   k3b    : q/k/v, attn softmax, s_att, s_out -> stored transposed bf16
//   k4     : out[n][h*64+e] = sum_m w[h,n,m]*s_out[h,m,e]  (WMMA bf16)
// HBM traffic ~1.0 GB -> ~44 us at 23.3 TB/s; compute is far below WMMA peak.

typedef __attribute__((ext_vector_type(16))) __bf16 v16bf;
typedef __attribute__((ext_vector_type(8)))  float  v8f;

#define HEADS 8
#define MSL   64
#define DIMC  512
#define DH    64
#define NPB   32      // nodes per block in k1/k4 (2 row tiles of 16)
#define NK    1024    // split-K node chunk in k2
#define LSTR  130     // padded LDS row stride (65 dwords -> conflict-free)

__device__ __forceinline__ v8f wmma_bf16(v16bf a, v16bf b, v8f c) {
  // D = A(16x32) * B(32x16) + C, f32 accumulate
  return __builtin_amdgcn_wmma_f32_16x16x32_bf16(
      /*neg_a=*/false, a, /*neg_b=*/false, b,
      /*c_mod=*/(short)0, c, /*reuse_a=*/false, /*reuse_b=*/false);
}

// ---------------------------------------------------------------- k_init ---
// zero accumulators + convert w_slice (512x512) to bf16.
__global__ __launch_bounds__(256) void k_init(float* __restrict__ d_g,
                                              float* __restrict__ s_raw,
                                              const float* __restrict__ w_slice,
                                              __bf16* __restrict__ wsl_bf) {
  int i = blockIdx.x * 256 + threadIdx.x;
  if (i < HEADS * MSL) d_g[i] = 0.f;
  if (i < HEADS * MSL * DIMC) {          // 262144 == 512*512 as well
    s_raw[i] = 0.f;
    wsl_bf[i] = (__bf16)w_slice[i];
  }
}

// ------------------------------------------------------------------- k1 ----
// grid = N/32 blocks, 256 threads (8 waves, one head per wave).
__global__ __launch_bounds__(256) void k1_slice_softmax(
    const float* __restrict__ x, const __bf16* __restrict__ wsl_bf,
    const float* __restrict__ b_slice, const float* __restrict__ temperature,
    __bf16* __restrict__ w_g, float* __restrict__ d_g, int N) {
  const int node0 = blockIdx.x * NPB;
  const int h    = threadIdx.x >> 5;
  const int lane = threadIdx.x & 31;
  const int l16  = lane & 15;
  const int half = lane >> 4;

  v8f acc[2][4];
#pragma unroll
  for (int rt = 0; rt < 2; ++rt)
#pragma unroll
    for (int ct = 0; ct < 4; ++ct)
#pragma unroll
      for (int v = 0; v < 8; ++v) acc[rt][ct][v] = 0.f;

  for (int kk = 0; kk < DIMC; kk += 32) {
    v16bf a[2];
#pragma unroll
    for (int rt = 0; rt < 2; ++rt) {
      // A layout (16x32 bf16): lanes0-15 K{0-7,16-23}, lanes16-31 K{8-15,24-31}
      const float* p = x + (size_t)(node0 + rt * 16 + l16) * DIMC + kk + half * 8;
#pragma unroll
      for (int i = 0; i < 8; ++i) {
        a[rt][i]     = (__bf16)p[i];
        a[rt][i + 8] = (__bf16)p[i + 16];
      }
    }
#pragma unroll
    for (int ct = 0; ct < 4; ++ct) {
      // B layout (32x16 bf16): lane = column, lanes0-15 K{0-7,16-23}, lanes16-31 K{8-15,24-31}
      const __bf16* q =
          wsl_bf + (size_t)(h * MSL + ct * 16 + l16) * DIMC + kk + half * 8;
      v16bf b;
#pragma unroll
      for (int i = 0; i < 8; ++i) { b[i] = q[i]; b[i + 8] = q[i + 16]; }
#pragma unroll
      for (int rt = 0; rt < 2; ++rt)
        acc[rt][ct] = wmma_bf16(a[rt], b, acc[rt][ct]);
    }
  }

  float tmp = temperature[h];
  tmp = fminf(fmaxf(tmp, 0.1f), 5.0f);
  const float itemp = 1.0f / tmp;

  float bias[4];
#pragma unroll
  for (int ct = 0; ct < 4; ++ct) bias[ct] = b_slice[h * MSL + ct * 16 + l16];

  // softmax across the 64 M-columns (4 fragments x 16 lanes of a half-wave)
#pragma unroll
  for (int rt = 0; rt < 2; ++rt)
#pragma unroll
    for (int v = 0; v < 8; ++v) {
      float mx = -3.4e38f;
#pragma unroll
      for (int ct = 0; ct < 4; ++ct) {
        float val = (acc[rt][ct][v] + bias[ct]) * itemp;
        acc[rt][ct][v] = val;
        mx = fmaxf(mx, val);
      }
      mx = fmaxf(mx, __shfl_xor(mx, 1, 32));
      mx = fmaxf(mx, __shfl_xor(mx, 2, 32));
      mx = fmaxf(mx, __shfl_xor(mx, 4, 32));
      mx = fmaxf(mx, __shfl_xor(mx, 8, 32));
      float sum = 0.f;
#pragma unroll
      for (int ct = 0; ct < 4; ++ct) {
        float e = __expf(acc[rt][ct][v] - mx);
        acc[rt][ct][v] = e;
        sum += e;
      }
      sum += __shfl_xor(sum, 1, 32);
      sum += __shfl_xor(sum, 2, 32);
      sum += __shfl_xor(sum, 4, 32);
      sum += __shfl_xor(sum, 8, 32);
      float inv = 1.0f / sum;
#pragma unroll
      for (int ct = 0; ct < 4; ++ct) acc[rt][ct][v] *= inv;
    }

  // d[h][m] += sum over rows
#pragma unroll
  for (int ct = 0; ct < 4; ++ct) {
    float sm = 0.f;
#pragma unroll
    for (int rt = 0; rt < 2; ++rt)
#pragma unroll
      for (int v = 0; v < 8; ++v) sm += acc[rt][ct][v];
    sm += __shfl_xor(sm, 16, 32);
    if (half == 0) atomicAdd(&d_g[h * MSL + ct * 16 + l16], sm);
  }

  // store softmax weights bf16: w_g[n][h*64 + m]
#pragma unroll
  for (int rt = 0; rt < 2; ++rt)
#pragma unroll
    for (int ct = 0; ct < 4; ++ct) {
      int col = h * MSL + ct * 16 + l16;
#pragma unroll
      for (int v = 0; v < 8; ++v) {
        int row = node0 + rt * 16 + v + half * 8;  // C/D layout row
        w_g[(size_t)row * DIMC + col] = (__bf16)acc[rt][ct][v];
      }
    }
}

// ------------------------------------------------------------------- k2 ----
// grid = (HEADS * 8 c-blocks, nchunks); head = fastest dim for L2 reuse.
// s_raw[h][m][c] += sum_n w[h,n,m]*x[n,c] via LDS-transposed bf16 tiles.
__global__ __launch_bounds__(256) void k2_sraw(
    const float* __restrict__ x, const __bf16* __restrict__ w_g,
    float* __restrict__ s_raw, int N) {
  const int h     = blockIdx.x & (HEADS - 1);       // fastest -> L2 tile reuse
  const int cbase = (blockIdx.x >> 3) * 64;
  const int nbeg  = blockIdx.y * NK;
  const int nend  = (nbeg + NK < N) ? (nbeg + NK) : N;

  __shared__ __bf16 wT[64 * LSTR];  // [m][n], padded stride
  __shared__ __bf16 xT[64 * LSTR];  // [c][n], padded stride

  const int wave = threadIdx.x >> 5;
  const int lane = threadIdx.x & 31;
  const int l16  = lane & 15;
  const int half = lane >> 4;
  const int mt   = wave >> 1;        // m tile 0..3
  const int ct0  = (wave & 1) * 2;   // c tiles {ct0, ct0+1}

  v8f acc[2];
#pragma unroll
  for (int j = 0; j < 2; ++j)
#pragma unroll
    for (int v = 0; v < 8; ++v) acc[j][v] = 0.f;

  for (int n0 = nbeg; n0 < nend; n0 += 128) {
    // prefetch next stage's rows into L2/WGP$ while this one is consumed
    if (threadIdx.x < 128) {
      int pn = n0 + 128 + threadIdx.x;
      if (pn < nend) {
        __builtin_prefetch(&w_g[(size_t)pn * DIMC + h * MSL], 0, 1);
        __builtin_prefetch(&x[(size_t)pn * DIMC + cbase], 0, 1);
      }
    }
    // stage 128 nodes, transposed, zero-padded past nend
    for (int i = threadIdx.x; i < 64 * 128; i += 256) {
      int mm = i & 63, nn = i >> 6;
      int node = n0 + nn;
      __bf16 wv = (__bf16)0.f;
      __bf16 xv = (__bf16)0.f;
      if (node < nend) {
        wv = w_g[(size_t)node * DIMC + h * MSL + mm];
        xv = (__bf16)x[(size_t)node * DIMC + cbase + mm];
      }
      wT[mm * LSTR + nn] = wv;
      xT[mm * LSTR + nn] = xv;
    }
    __syncthreads();

#pragma unroll
    for (int ks = 0; ks < 4; ++ks) {
      const __bf16* pa = &wT[(mt * 16 + l16) * LSTR + ks * 32 + half * 8];
      v16bf a;
#pragma unroll
      for (int i = 0; i < 8; ++i) { a[i] = pa[i]; a[i + 8] = pa[i + 16]; }
#pragma unroll
      for (int j = 0; j < 2; ++j) {
        const __bf16* pb =
            &xT[((ct0 + j) * 16 + l16) * LSTR + ks * 32 + half * 8];
        v16bf b;
#pragma unroll
        for (int i = 0; i < 8; ++i) { b[i] = pb[i]; b[i + 8] = pb[i + 16]; }
        acc[j] = wmma_bf16(a, b, acc[j]);
      }
    }
    __syncthreads();
  }

#pragma unroll
  for (int j = 0; j < 2; ++j)
#pragma unroll
    for (int v = 0; v < 8; ++v) {
      int m  = mt * 16 + v + half * 8;
      int cc = cbase + (ct0 + j) * 16 + l16;
      atomicAdd(&s_raw[(size_t)(h * MSL + m) * DIMC + cc], acc[j][v]);
    }
}

// ------------------------------------------------------------------ k3a ----
// s_mid[h][m][dh] = sum_c (s_raw[h][m][c]/(d+1e-5)) * w1[dh][c] + b1[dh]
__global__ __launch_bounds__(256) void k3a_project(
    const float* __restrict__ s_raw, const float* __restrict__ d_g,
    const float* __restrict__ w1, const float* __restrict__ b1,
    float* __restrict__ s_mid) {
  int idx = blockIdx.x * 256 + threadIdx.x;  // HEADS*MSL*DH = 32768
  int dh = idx & 63;
  int m  = (idx >> 6) & 63;
  int h  = idx >> 12;
  const float* sr = s_raw + (size_t)(h * MSL + m) * DIMC;
  const float* wr = w1 + (size_t)dh * DIMC;
  float acc = 0.f;
  for (int c = 0; c < DIMC; ++c) acc += sr[c] * wr[c];
  float dn = 1.0f / (d_g[h * MSL + m] + 1e-5f);
  s_mid[idx] = acc * dn + b1[dh];
}

// ------------------------------------------------------------------ k3b ----
// per-head: q/k/v, attn softmax, s_att, s_out -> transposed bf16 [h][e][m]
__global__ __launch_bounds__(256) void k3b_attention(
    const float* __restrict__ s_mid, const float* __restrict__ wq,
    const float* __restrict__ wk, const float* __restrict__ wv,
    const float* __restrict__ w3, const float* __restrict__ b3,
    __bf16* __restrict__ s_outT) {
  __shared__ float qbuf[64 * 64];  // q, later v
  __shared__ float kbuf[64 * 64];  // k, later s_att
  __shared__ float abuf[64 * 64];  // attn
  const int h = blockIdx.x;
  const int t = threadIdx.x;
  const float* s = s_mid + (size_t)h * MSL * DH;

  // q, k
#pragma unroll 4
  for (int i = 0; i < 16; ++i) {
    int o = t * 16 + i;
    int m = o >> 6, e = o & 63;
    float aq = 0.f, ak = 0.f;
    for (int j = 0; j < DH; ++j) {
      float sv = s[m * DH + j];
      aq += sv * wq[e * DH + j];
      ak += sv * wk[e * DH + j];
    }
    qbuf[o] = aq;
    kbuf[o] = ak;
  }
  __syncthreads();

  // attn logits + softmax (4 threads per row, shfl within groups of 4)
  const int r = t >> 2, c0 = (t & 3) * 16;
  float att[16];
  float mx = -3.4e38f;
#pragma unroll 4
  for (int i = 0; i < 16; ++i) {
    float a = 0.f;
    for (int e = 0; e < DH; ++e) a += qbuf[r * 64 + e] * kbuf[(c0 + i) * 64 + e];
    a *= 0.125f;  // 1/sqrt(64)
    att[i] = a;
    mx = fmaxf(mx, a);
  }
  mx = fmaxf(mx, __shfl_xor(mx, 1, 32));
  mx = fmaxf(mx, __shfl_xor(mx, 2, 32));
  float sum = 0.f;
#pragma unroll
  for (int i = 0; i < 16; ++i) { att[i] = __expf(att[i] - mx); sum += att[i]; }
  sum += __shfl_xor(sum, 1, 32);
  sum += __shfl_xor(sum, 2, 32);
  float inv = 1.0f / sum;
#pragma unroll
  for (int i = 0; i < 16; ++i) att[i] *= inv;
  __syncthreads();  // all q/k reads done

  // attn -> abuf; v -> qbuf
#pragma unroll
  for (int i = 0; i < 16; ++i) abuf[r * 64 + c0 + i] = att[i];
#pragma unroll 4
  for (int i = 0; i < 16; ++i) {
    int o = t * 16 + i;
    int m = o >> 6, e = o & 63;
    float av = 0.f;
    for (int j = 0; j < DH; ++j) av += s[m * DH + j] * wv[e * DH + j];
    qbuf[o] = av;
  }
  __syncthreads();

  // s_att = attn @ v -> kbuf
#pragma unroll 4
  for (int i = 0; i < 16; ++i) {
    int o = t * 16 + i;
    int m = o >> 6, e = o & 63;
    float acc = 0.f;
    for (int j = 0; j < 64; ++j) acc += abuf[m * 64 + j] * qbuf[j * 64 + e];
    kbuf[o] = acc;
  }
  __syncthreads();

  // s_out = s_att @ w3^T + b3, stored transposed bf16 for k4's B operand
#pragma unroll 4
  for (int i = 0; i < 16; ++i) {
    int o = t * 16 + i;
    int m = o >> 6, dd = o & 63;
    float acc = b3[dd];
    for (int e = 0; e < DH; ++e) acc += kbuf[m * 64 + e] * w3[dd * DH + e];
    s_outT[(size_t)(h * DH + dd) * MSL + m] = (__bf16)acc;
  }
}

// ------------------------------------------------------------------- k4 ----
// out[n][h*64+e] = sum_m w[h,n,m] * s_out[h,m,e]
__global__ __launch_bounds__(256) void k4_deslice(
    const __bf16* __restrict__ w_g, const __bf16* __restrict__ s_outT,
    float* __restrict__ out, int N) {
  const int node0 = blockIdx.x * NPB;
  const int h    = threadIdx.x >> 5;
  const int lane = threadIdx.x & 31;
  const int l16  = lane & 15;
  const int half = lane >> 4;

  v8f acc[2][4];
#pragma unroll
  for (int rt = 0; rt < 2; ++rt)
#pragma unroll
    for (int ct = 0; ct < 4; ++ct)
#pragma unroll
      for (int v = 0; v < 8; ++v) acc[rt][ct][v] = 0.f;

#pragma unroll
  for (int kk = 0; kk < MSL; kk += 32) {
    v16bf a[2];
#pragma unroll
    for (int rt = 0; rt < 2; ++rt) {
      const __bf16* p =
          w_g + (size_t)(node0 + rt * 16 + l16) * DIMC + h * MSL + kk + half * 8;
#pragma unroll
      for (int i = 0; i < 8; ++i) { a[rt][i] = p[i]; a[rt][i + 8] = p[i + 16]; }
    }
#pragma unroll
    for (int ct = 0; ct < 4; ++ct) {
      const __bf16* q =
          s_outT + (size_t)(h * DH + ct * 16 + l16) * MSL + kk + half * 8;
      v16bf b;
#pragma unroll
      for (int i = 0; i < 8; ++i) { b[i] = q[i]; b[i + 8] = q[i + 16]; }
#pragma unroll
      for (int rt = 0; rt < 2; ++rt)
        acc[rt][ct] = wmma_bf16(a[rt], b, acc[rt][ct]);
    }
  }

#pragma unroll
  for (int rt = 0; rt < 2; ++rt)
#pragma unroll
    for (int ct = 0; ct < 4; ++ct) {
      int col = h * MSL + ct * 16 + l16;
#pragma unroll
      for (int v = 0; v < 8; ++v) {
        int row = node0 + rt * 16 + v + half * 8;
        out[(size_t)row * DIMC + col] = acc[rt][ct][v];
      }
    }
}

// ---------------------------------------------------------------- launch ---
extern "C" void kernel_launch(void* const* d_in, const int* in_sizes, int n_in,
                              void* d_out, int out_size, void* d_ws,
                              size_t ws_size, hipStream_t stream) {
  const float* x           = (const float*)d_in[0];
  const float* temperature = (const float*)d_in[1];
  const float* w_slice     = (const float*)d_in[2];
  const float* b_slice     = (const float*)d_in[3];
  const float* w1          = (const float*)d_in[4];
  const float* b1          = (const float*)d_in[5];
  const float* wq          = (const float*)d_in[6];
  const float* wk          = (const float*)d_in[7];
  const float* wv          = (const float*)d_in[8];
  const float* w3          = (const float*)d_in[9];
  const float* b3          = (const float*)d_in[10];
  float* out = (float*)d_out;
  const int N = in_sizes[0] / DIMC;  // 100000

  // workspace carve-up (~105 MB)
  char* ws = (char*)d_ws;
  __bf16* w_g = (__bf16*)ws;  // N*512 bf16
  size_t off = (((size_t)N * DIMC * sizeof(__bf16)) + 255) & ~(size_t)255;
  float* s_raw = (float*)(ws + off);
  off += (size_t)HEADS * MSL * DIMC * sizeof(float);
  float* d_g = (float*)(ws + off);
  off += (size_t)HEADS * MSL * sizeof(float);
  off = (off + 255) & ~(size_t)255;
  float* s_mid = (float*)(ws + off);
  off += (size_t)HEADS * MSL * DH * sizeof(float);
  __bf16* s_outT = (__bf16*)(ws + off);
  off += (size_t)HEADS * MSL * DH * sizeof(__bf16);
  off = (off + 255) & ~(size_t)255;
  __bf16* wsl_bf = (__bf16*)(ws + off);  // 512*512 bf16 (L2-resident)

  k_init<<<dim3((HEADS * MSL * DIMC + 255) / 256), dim3(256), 0, stream>>>(
      d_g, s_raw, w_slice, wsl_bf);
  k1_slice_softmax<<<dim3(N / NPB), dim3(256), 0, stream>>>(
      x, wsl_bf, b_slice, temperature, w_g, d_g, N);
  const int nchunks = (N + NK - 1) / NK;
  k2_sraw<<<dim3(HEADS * (DIMC / 64), nchunks), dim3(256), 0, stream>>>(
      x, w_g, s_raw, N);
  k3a_project<<<dim3(HEADS * MSL * DH / 256), dim3(256), 0, stream>>>(
      s_raw, d_g, w1, b1, s_mid);
  k3b_attention<<<dim3(HEADS), dim3(256), 0, stream>>>(
      s_mid, wq, wk, wv, w3, b3, s_outT);
  k4_deslice<<<dim3(N / NPB), dim3(256), 0, stream>>>(w_g, s_outT, out, N);
}
